// MyRnn2_43413529428168
// MI455X (gfx1250) — matrix operand order repaired and checked by reading, not compiled, not run on hardware
//
#include <hip/hip_runtime.h>
#include <hip/hip_bf16.h>

typedef __attribute__((ext_vector_type(16))) _Float16 v16h;
typedef __attribute__((ext_vector_type(8)))  float    v8f;
typedef _Float16 half_t;

#define HID   128
#define NBAT  16384
#define TSTEP 28
#define IN0   28
#define NBROW 32               // batch rows per block (2 row-tiles of 16)

// Pre-swizzled weight fragments (B-matrix layout for v_wmma_f32_16x16x32_f16):
// one fragment = 32 lanes x 16 halfs = 512 halfs = 1KB, covering a 32(K) x 16(N) tile.
//   lane = n_local + 16*((k_local>>3)&1), half j = 8*(k_local>>4) + (k_local&7)
#define F_WIH1 24              // 24 ntiles, K=28 padded to 32
#define F_WHH1 96              // 24 ntiles x 4 ktiles (K=128)
#define F_WIH2 96              // 24 ntiles x 4 ktiles
#define F_WOUT 4               // 1 ntile (10 rows padded to 16) x 4 ktiles
#define F_TOTAL (F_WIH1 + F_WHH1 + F_WIH2 + F_WOUT)   // 220 frags = 220KB

#define W_BYTES   (F_TOTAL * 1024)             // 225280 B of weights in LDS
#define SG_HALFO  (F_TOTAL * 512)              // half-offset of sG region in dyn LDS
#define DYN_BYTES (W_BYTES + NBROW * 512 * 4)  // + 64KB f32 gate buffer

__device__ __forceinline__ float sigm(float v) {
  return 1.0f / (1.0f + __expf(-v));
}
__device__ __forceinline__ float tanh_fast(float v) {
  // tanh(x) = 1 - 2/(1 + e^{2x}); one v_exp_f32 (TRANS, co-executes with WMMA)
  return 1.0f - 2.0f / (1.0f + __expf(2.0f * v));
}

__device__ __forceinline__ v8f wmma_f16(v16h a, v16h b, v8f c) {
  return __builtin_amdgcn_wmma_f32_16x16x32_f16(false, a, false, b,
                                                (short)0, c, false, false);
}

// ---------------------------------------------------------------------------
// Kernel 1: swizzle f32 weights into f16 B-fragment order in workspace.
// ---------------------------------------------------------------------------
__global__ void prep_weights(const float* __restrict__ Wih1,
                             const float* __restrict__ Whh1,
                             const float* __restrict__ Wih2,
                             const float* __restrict__ Wout,
                             half_t* __restrict__ wsH) {
  int t = blockIdx.x * blockDim.x + threadIdx.x;
  const int total = F_TOTAL * 512;
  if (t >= total) return;
  int f    = t >> 9;
  int idx  = t & 511;
  int lane = idx >> 4;
  int j    = idx & 15;
  int nl   = lane & 15;                 // N within tile
  int kl   = ((j >> 3) << 4) | ((lane >> 4) << 3) | (j & 7);    // K within tile (0..31)
  float v = 0.0f;
  if (f < F_WIH1) {                                   // Wih1 [384 x 28], pad K
    int n = f * 16 + nl;
    if (kl < IN0) v = Wih1[n * IN0 + kl];
  } else if (f < F_WIH1 + F_WHH1) {                   // Whh1 [384 x 128]
    int f2 = f - F_WIH1;
    int nt = f2 >> 2, kt = f2 & 3;
    v = Whh1[(nt * 16 + nl) * HID + kt * 32 + kl];
  } else if (f < F_WIH1 + F_WHH1 + F_WIH2) {          // Wih2 [384 x 128]
    int f2 = f - (F_WIH1 + F_WHH1);
    int nt = f2 >> 2, kt = f2 & 3;
    v = Wih2[(nt * 16 + nl) * HID + kt * 32 + kl];
  } else {                                            // Wout [10 x 128], pad N
    int kt = f - (F_WIH1 + F_WHH1 + F_WIH2);
    if (nl < 10) v = Wout[nl * HID + kt * 32 + kl];
  }
  wsH[t] = (half_t)v;
}

// ---------------------------------------------------------------------------
// Kernel 2: 28-step 2-layer GRU. One block = 32 batch rows (2 row-tiles),
// 8 waves. All weights async-copied to LDS once; B fragments for each gate
// tile are preloaded as a batch (clause-able ds_loads, partial waits) and
// each B fragment feeds TWO independent WMMA accumulator chains.
// ---------------------------------------------------------------------------
__launch_bounds__(256)
__global__ void gru_main(const float* __restrict__ x,
                         const float* __restrict__ bih1, const float* __restrict__ bhh1,
                         const float* __restrict__ bih2, const float* __restrict__ bhh2,
                         const float* __restrict__ bout,
                         const half_t* __restrict__ wsH,
                         float* __restrict__ out) {
  extern __shared__ __align__(16) half_t sW[];        // [0,220KB): weights, then 64KB sG
  float* sGp = (float*)(sW + SG_HALFO);               // gates: [32 rows][512 cols] f32
  // sG columns: [0,256) combined r|z, [256,384) ic, [384,512) hc

  __shared__ __align__(32) half_t sH1frag[4][2][32][16];  // h1, A-frag layout, 8 KB
  __shared__ __align__(32) half_t sH0frag[4][2][32][16];  // h0, A-frag layout, 8 KB
  __shared__ float sB[1024];  // biases: [0,256) rz1 sum, [256,384) ic1, [384,512) hc1,
                              //         [512,768) rz2 sum, [768,896) c2, [896,1024) hc2

  const int tid  = threadIdx.x;
  const int lane = tid & 31;
  const int w    = tid >> 5;
  const int nlo  = lane & 15;
  const int hi   = lane >> 4;
  const int row0 = blockIdx.x * NBROW;

  // ---- stage biases (pre-summed where the math allows) ----
  sB[tid]       = bih1[tid] + bhh1[tid];        // tid in [0,256): r|z cell1
  sB[512 + tid] = bih2[tid] + bhh2[tid];        // r|z cell2
  if (tid < 128) {
    sB[256 + tid] = bih1[256 + tid];            // ic1
    sB[384 + tid] = bhh1[256 + tid];            // hc1
    sB[768 + tid] = bih2[256 + tid];            // ic2
    sB[896 + tid] = bhh2[256 + tid];            // hc2 (constant: cell2 h == 0)
  }

  // ---- h1 starts at zero ----
  for (int i = tid; i < 4 * 2 * 32 * 16; i += 256) ((half_t*)sH1frag)[i] = (half_t)0.0f;

  // ---- async-copy all weight fragments into LDS (CDNA5 async path) ----
  {
    const unsigned ldsBase = (unsigned)(unsigned long long)(uintptr_t)(&sW[0]);
    for (int i = tid; i < F_TOTAL * 64; i += 256) {   // 14080 x 16B chunks
      unsigned long long g = (unsigned long long)(uintptr_t)(wsH + (size_t)i * 8);
      unsigned l = ldsBase + (unsigned)i * 16u;
      asm volatile("global_load_async_to_lds_b128 %0, %1, off"
                   :: "v"(l), "v"(g) : "memory");
    }
    asm volatile("s_wait_asynccnt 0x0" ::: "memory");
  }
  __syncthreads();

  const half_t* sWih1 = sW;
  const half_t* sWhh1 = sW + F_WIH1 * 512;
  const half_t* sWih2 = sW + (F_WIH1 + F_WHH1) * 512;
  const half_t* sWout = sW + (F_WIH1 + F_WHH1 + F_WIH2) * 512;

  for (int t = 0; t < TSTEP; ++t) {
    __syncthreads();   // sH1frag ready, sG free

    // ---- x_t A-fragments for both row-tiles (f32 -> f16 per lane) ----
    // A layout: lane m = lane&15; lo lanes: j0..7=K0..7, j8..15=K16..23
    //           hi lanes: j0..7=K8..15, j8..11=K24..27, j12..15=0
    v16h ax[2];
#pragma unroll
    for (int rt = 0; rt < 2; ++rt) {
      const float* xr = x + (size_t)(row0 + rt * 16 + nlo) * 784 + t * 28;
      if (t + 1 < TSTEP) __builtin_prefetch(xr + 28, 0, 0);  // global_prefetch_b8
      float4 a0, a1, a2, a3;
      if (hi == 0) {
        a0 = *(const float4*)(xr + 0);
        a1 = *(const float4*)(xr + 4);
        a2 = *(const float4*)(xr + 16);
        a3 = *(const float4*)(xr + 20);
      } else {
        a0 = *(const float4*)(xr + 8);
        a1 = *(const float4*)(xr + 12);
        a2 = *(const float4*)(xr + 24);
        a3 = make_float4(0.f, 0.f, 0.f, 0.f);
      }
      v16h a;
      a[0]  = (_Float16)a0.x; a[1]  = (_Float16)a0.y;
      a[2]  = (_Float16)a0.z; a[3]  = (_Float16)a0.w;
      a[4]  = (_Float16)a1.x; a[5]  = (_Float16)a1.y;
      a[6]  = (_Float16)a1.z; a[7]  = (_Float16)a1.w;
      a[8]  = (_Float16)a2.x; a[9]  = (_Float16)a2.y;
      a[10] = (_Float16)a2.z; a[11] = (_Float16)a2.w;
      a[12] = (_Float16)a3.x; a[13] = (_Float16)a3.y;
      a[14] = (_Float16)a3.z; a[15] = (_Float16)a3.w;
      ax[rt] = a;
    }

    // ---- h1 A-fragments ----
    v16h ah[2][4];
#pragma unroll
    for (int rt = 0; rt < 2; ++rt)
#pragma unroll
      for (int kt = 0; kt < 4; ++kt)
        ah[rt][kt] = *(const v16h*)(&sH1frag[kt][rt][lane][0]);

    // ---- phase 1: uniform per-wave tile mix (no divergent WMMA paths) ----
    // (a) two combined r|z tiles: gi1+gh1 accumulated together.
    //     All 5 B fragments preloaded so ds_loads clause and waits overlap
    //     the 10-WMMA chain.
#pragma unroll
    for (int i = 0; i < 2; ++i) {
      const int q = 2 * w + i;
      const half_t* bp = sWhh1 + (size_t)(q * 4) * 512 + lane * 16;
      v16h b0 = *(const v16h*)(bp + 0 * 512);
      v16h b1 = *(const v16h*)(bp + 1 * 512);
      v16h b2 = *(const v16h*)(bp + 2 * 512);
      v16h b3 = *(const v16h*)(bp + 3 * 512);
      v16h b4 = *(const v16h*)(sWih1 + (size_t)q * 512 + lane * 16);
      v8f acc0 = {}, acc1 = {};
      acc0 = wmma_f16(ah[0][0], b0, acc0);  acc1 = wmma_f16(ah[1][0], b0, acc1);
      acc0 = wmma_f16(ah[0][1], b1, acc0);  acc1 = wmma_f16(ah[1][1], b1, acc1);
      acc0 = wmma_f16(ah[0][2], b2, acc0);  acc1 = wmma_f16(ah[1][2], b2, acc1);
      acc0 = wmma_f16(ah[0][3], b3, acc0);  acc1 = wmma_f16(ah[1][3], b3, acc1);
      acc0 = wmma_f16(ax[0],    b4, acc0);  acc1 = wmma_f16(ax[1],    b4, acc1);
#pragma unroll
      for (int e = 0; e < 8; ++e) {
        sGp[(size_t)(hi * 8 + e) * 512 + q * 16 + nlo]      = acc0[e];
        sGp[(size_t)(16 + hi * 8 + e) * 512 + q * 16 + nlo] = acc1[e];
      }
    }
    // (b) one ic tile: x @ Wih1^T only
    {
      v16h bx = *(const v16h*)(sWih1 + (size_t)(16 + w) * 512 + lane * 16);
      v8f acc0 = {}, acc1 = {};
      acc0 = wmma_f16(ax[0], bx, acc0);
      acc1 = wmma_f16(ax[1], bx, acc1);
#pragma unroll
      for (int e = 0; e < 8; ++e) {
        sGp[(size_t)(hi * 8 + e) * 512 + 256 + w * 16 + nlo]      = acc0[e];
        sGp[(size_t)(16 + hi * 8 + e) * 512 + 256 + w * 16 + nlo] = acc1[e];
      }
    }
    // (c) one hc tile: h1 @ Whh1^T only (4 B frags preloaded)
    {
      const half_t* bp = sWhh1 + (size_t)((16 + w) * 4) * 512 + lane * 16;
      v16h b0 = *(const v16h*)(bp + 0 * 512);
      v16h b1 = *(const v16h*)(bp + 1 * 512);
      v16h b2 = *(const v16h*)(bp + 2 * 512);
      v16h b3 = *(const v16h*)(bp + 3 * 512);
      v8f acc0 = {}, acc1 = {};
      acc0 = wmma_f16(ah[0][0], b0, acc0);  acc1 = wmma_f16(ah[1][0], b0, acc1);
      acc0 = wmma_f16(ah[0][1], b1, acc0);  acc1 = wmma_f16(ah[1][1], b1, acc1);
      acc0 = wmma_f16(ah[0][2], b2, acc0);  acc1 = wmma_f16(ah[1][2], b2, acc1);
      acc0 = wmma_f16(ah[0][3], b3, acc0);  acc1 = wmma_f16(ah[1][3], b3, acc1);
#pragma unroll
      for (int e = 0; e < 8; ++e) {
        sGp[(size_t)(hi * 8 + e) * 512 + 384 + w * 16 + nlo]      = acc0[e];
        sGp[(size_t)(16 + hi * 8 + e) * 512 + 384 + w * 16 + nlo] = acc1[e];
      }
    }
    __syncthreads();

    // ---- elementwise cell 1: h0 = relu(gru(x_t, h1)) ----
#pragma unroll
    for (int u = 0; u < 16; ++u) {
      int lin = tid + u * 256;
      int m = lin >> 7, col = lin & 127;
      int kt = col >> 5, kl = col & 31;
      int g  = kl >> 3;
      int jj = ((g >> 1) << 3) | (kl & 7);
      int ln = (m & 15) + ((g & 1) << 4);
      int rt = m >> 4;
      float r  = sigm(sGp[(size_t)m * 512 + col] + sB[col]);
      float z  = sigm(sGp[(size_t)m * 512 + 128 + col] + sB[128 + col]);
      float hc = sGp[(size_t)m * 512 + 384 + col] + sB[384 + col];
      float c  = tanh_fast(sGp[(size_t)m * 512 + 256 + col] + sB[256 + col] + r * hc);
      float hprev = (float)sH1frag[kt][rt][ln][jj];
      float h0 = (1.0f - z) * c + z * hprev;
      h0 = fmaxf(h0, 0.0f);
      sH0frag[kt][rt][ln][jj] = (half_t)h0;
    }
    __syncthreads();

    // ---- phase 2: gi2 = h0 @ Wih2^T (24 tiles, 3 per wave, K=128) ----
    v16h a0f[2][4];
#pragma unroll
    for (int rt = 0; rt < 2; ++rt)
#pragma unroll
      for (int kt = 0; kt < 4; ++kt)
        a0f[rt][kt] = *(const v16h*)(&sH0frag[kt][rt][lane][0]);
#pragma unroll
    for (int i = 0; i < 3; ++i) {
      const int q = w * 3 + i;
      const half_t* bp = sWih2 + (size_t)(q * 4) * 512 + lane * 16;
      v16h b0 = *(const v16h*)(bp + 0 * 512);
      v16h b1 = *(const v16h*)(bp + 1 * 512);
      v16h b2 = *(const v16h*)(bp + 2 * 512);
      v16h b3 = *(const v16h*)(bp + 3 * 512);
      v8f acc0 = {}, acc1 = {};
      acc0 = wmma_f16(a0f[0][0], b0, acc0);  acc1 = wmma_f16(a0f[1][0], b0, acc1);
      acc0 = wmma_f16(a0f[0][1], b1, acc0);  acc1 = wmma_f16(a0f[1][1], b1, acc1);
      acc0 = wmma_f16(a0f[0][2], b2, acc0);  acc1 = wmma_f16(a0f[1][2], b2, acc1);
      acc0 = wmma_f16(a0f[0][3], b3, acc0);  acc1 = wmma_f16(a0f[1][3], b3, acc1);
#pragma unroll
      for (int e = 0; e < 8; ++e) {
        sGp[(size_t)(hi * 8 + e) * 512 + q * 16 + nlo]      = acc0[e];
        sGp[(size_t)(16 + hi * 8 + e) * 512 + q * 16 + nlo] = acc1[e];
      }
    }
    __syncthreads();

    // ---- elementwise cell 2 (h==0 => gh2 = bhh2): h1 = (1-z2)*c2 ----
#pragma unroll
    for (int u = 0; u < 16; ++u) {
      int lin = tid + u * 256;
      int m = lin >> 7, col = lin & 127;
      float r2 = sigm(sGp[(size_t)m * 512 + col] + sB[512 + col]);
      float z2 = sigm(sGp[(size_t)m * 512 + 128 + col] + sB[640 + col]);
      float c2 = tanh_fast(sGp[(size_t)m * 512 + 256 + col] + sB[768 + col] + r2 * sB[896 + col]);
      float h1v = (1.0f - z2) * c2;
      int kt = col >> 5, kl = col & 31;
      int g  = kl >> 3;
      int jj = ((g >> 1) << 3) | (kl & 7);
      int ln = (m & 15) + ((g & 1) << 4);
      sH1frag[kt][m >> 4][ln][jj] = (half_t)h1v;
    }
  }
  __syncthreads();

  // ---- output projection: h1 @ Wout^T + bout (waves 0,1; N padded 10->16) ----
  if (w < 2) {
    const int rt = w;
    v8f acc = {};
#pragma unroll
    for (int kt = 0; kt < 4; ++kt) {
      v16h a = *(const v16h*)(&sH1frag[kt][rt][lane][0]);
      v16h b = *(const v16h*)(sWout + (size_t)kt * 512 + lane * 16);
      acc = wmma_f16(a, b, acc);
    }
    if (nlo < 10) {
      float bb = bout[nlo];
#pragma unroll
      for (int e = 0; e < 8; ++e) {
        int m = rt * 16 + hi * 8 + e;
        out[(size_t)(row0 + m) * 10 + nlo] = acc[e] + bb;
      }
    }
  }
}

// ---------------------------------------------------------------------------
extern "C" void kernel_launch(void* const* d_in, const int* in_sizes, int n_in,
                              void* d_out, int out_size, void* d_ws, size_t ws_size,
                              hipStream_t stream) {
  const float* x    = (const float*)d_in[0];
  const float* Wih1 = (const float*)d_in[1];
  const float* Whh1 = (const float*)d_in[2];
  const float* bih1 = (const float*)d_in[3];
  const float* bhh1 = (const float*)d_in[4];
  const float* Wih2 = (const float*)d_in[5];
  // d_in[6] = Whh2 is mathematically unused (cell 2 hidden state is always 0)
  const float* bih2 = (const float*)d_in[7];
  const float* bhh2 = (const float*)d_in[8];
  const float* Wout = (const float*)d_in[9];
  const float* bout = (const float*)d_in[10];

  half_t* wsH = (half_t*)d_ws;   // 220 KB of f16 weight fragments

  const int totalH = F_TOTAL * 512;
  prep_weights<<<(totalH + 255) / 256, 256, 0, stream>>>(Wih1, Whh1, Wih2, Wout, wsH);

  gru_main<<<NBAT / NBROW, 256, DYN_BYTES, stream>>>(x, bih1, bhh1, bih2, bhh2, bout,
                                                     wsH, (float*)d_out);
}
